// EdgeConvolutionLayer_13331578486913
// MI455X (gfx1250) — compile-verified, matrix operand order checked by reading, not run on hardware
//
#include <hip/hip_runtime.h>

// ---------------------------------------------------------------------------
// EdgeConv (B=256,P=128,C=2,F=7,K=16,H=128,O=256) for gfx1250 (MI455X).
// Compute-bound on GEMM2 (34.4 GFLOP) -> v_wmma_f32_16x16x32_f16.
// One wave per point; GEMM1 computed transposed so h1 writeback is b128;
// independent accumulators / paired tiles to hide WMMA->VALU hazards.
// ---------------------------------------------------------------------------

typedef __attribute__((ext_vector_type(16))) _Float16 v16h;
typedef __attribute__((ext_vector_type(8)))  _Float16 v8h;
typedef __attribute__((ext_vector_type(8)))  float    v8f;

#define B_   256
#define P_   128
#define C_   2
#define F_   7
#define K_   16
#define H_   128
#define O_   256
#define E_   18        // 2*(C+F)
#define EP_  32        // E padded to WMMA K-chunk
#define CH_  (C_ + F_ + 1)   // 10 channels in x
#define OUTW (O_ + 1)        // 257
#define BIGF 1.0e9f

union V16 { v16h v; v8h h[2]; };
union H8  { v8h  v; _Float16 e[8]; };

__device__ __forceinline__ void wave_lds_sync() {
  // per-wave LDS producer->consumer: wait for our ds ops, fence scheduling
  asm volatile("s_wait_dscnt 0" ::: "memory");
  __builtin_amdgcn_wave_barrier();
}

// W2 (H x O fp32, row-major) -> W2^T (O x H f16) in workspace.
__global__ void prep_w2t_kernel(const float* __restrict__ W2,
                                _Float16* __restrict__ w2t) {
  int i = blockIdx.x * blockDim.x + threadIdx.x;   // i = n*H + k
  if (i < H_ * O_) {
    int n = i >> 7;          // output channel
    int k = i & (H_ - 1);    // hidden channel
    w2t[i] = (_Float16)W2[k * O_ + n];
  }
}

__global__ __launch_bounds__(256) void edgeconv_kernel(
    const float* __restrict__ x,       // (B,P,10)
    const float* __restrict__ W1,      // (E,H)
    const float* __restrict__ b1,      // (H)
    const float* __restrict__ b2,      // (O)
    const _Float16* __restrict__ w2t,  // (O,H) f16, from prep
    float* __restrict__ out) {         // (B,P,257)

  __shared__ float xs[P_][CH_];                                 //  5120 B
  __shared__ __align__(16) _Float16 w1t[H_][EP_];               //  8192 B
  __shared__ __align__(16) float b1s[H_];                       //   512 B
  __shared__ __align__(16) float b2s[O_];                       //  1024 B
  __shared__ unsigned short nbr[64][K_];                        //  2048 B
  __shared__ __align__(16) _Float16 edg[8][16][EP_];            //  8192 B
  __shared__ __align__(16) _Float16 h1s[8][16][H_];             // 32768 B

  const int tid = threadIdx.x;
  const int b   = blockIdx.x >> 1;
  const int p0  = (blockIdx.x & 1) * 64;   // this block handles 64 points

  // ---- Stage 0: load points, W1^T (padded f16), biases into LDS ---------
  for (int i = tid; i < P_ * CH_; i += 256)
    (&xs[0][0])[i] = x[(size_t)b * P_ * CH_ + i];
  for (int i = tid; i < H_ * EP_; i += 256) {
    int n = i >> 5, k = i & (EP_ - 1);
    w1t[n][k] = (k < E_) ? (_Float16)W1[k * H_ + n] : (_Float16)0.0f;
  }
  if (tid < H_) b1s[tid] = b1[tid];
  if (tid < O_) b2s[tid] = b2[tid];
  __syncthreads();

  // ---- Stage 1: kNN (64 threads, one per point) -------------------------
  if (tid < 64) {
    const int p = p0 + tid;
    const float px = xs[p][0], py = xs[p][1];
    float bd[K_ + 1];
    int   bi[K_ + 1];
#pragma unroll
    for (int s = 0; s <= K_; ++s) { bd[s] = 4.0f * BIGF; bi[s] = 0; }
    for (int j = 0; j < P_; ++j) {
      const float mj = xs[j][CH_ - 1];
      const float dx = xs[j][0] - px, dy = xs[j][1] - py;
      const float d  = (mj > 0.0f) ? (dx * dx + dy * dy) : BIGF;
      if (d < bd[K_]) { bd[K_] = d; bi[K_] = j; }
#pragma unroll
      for (int s = K_; s >= 1; --s) {   // bubble new entry up (stable: strict <)
        if (bd[s] < bd[s - 1]) {
          float td = bd[s]; bd[s] = bd[s - 1]; bd[s - 1] = td;
          int   ti = bi[s]; bi[s] = bi[s - 1]; bi[s - 1] = ti;
        }
      }
    }
#pragma unroll
    for (int s = 1; s <= K_; ++s) nbr[tid][s - 1] = (unsigned short)bi[s];
  }
  __syncthreads();

  // ---- Stage 2: per-point fused MLP via WMMA ----------------------------
  const int w     = tid >> 5;
  const int lane  = tid & 31;
  const int m     = lane & 15;             // row (A) or column (B/D) index
  const int kb    = (lane >> 4) << 3;      // K base: 0 (lanes 0-15) / 8 (16-31)
  const int rbase = (lane >> 4) << 3;      // D-matrix M base: 0 / 8

  for (int iter = 0; iter < 8; ++iter) {
    const int pl = iter * 8 + w;           // local point 0..63
    const int p  = p0 + pl;
    const float maskp = xs[p][CH_ - 1];
    const size_t obase = (size_t)(b * P_ + p) * OUTW;

    // build 16x18 edge tile (padded to 16x32) in f16, one neighbor per lane
    if (lane < 16) {
      const int nb = nbr[pl][lane];
      float e[EP_];
#pragma unroll
      for (int q = 0; q < EP_; ++q) e[q] = 0.0f;
      e[0] = xs[p][0]; e[1] = xs[p][1];
#pragma unroll
      for (int f = 0; f < F_; ++f) e[2 + f] = xs[p][2 + f];
      e[9]  = xs[nb][0] - e[0];
      e[10] = xs[nb][1] - e[1];
#pragma unroll
      for (int f = 0; f < F_; ++f) e[11 + f] = xs[nb][2 + f] - xs[p][2 + f];
#pragma unroll
      for (int q = 0; q < EP_; ++q) edg[w][lane][q] = (_Float16)e[q];
    }
    wave_lds_sync();

    // GEMM1 (transposed): h1^T = W1^T @ edge^T.
    // A = W1^T tile (rows = hidden), B = edge tile (cols = neighbors).
    // D: lane holds neighbor m, 8 consecutive hidden channels -> b128 store.
    V16 bedge;
    bedge.h[0] = *(const v8h*)&edg[w][m][kb];
    bedge.h[1] = *(const v8h*)&edg[w][m][kb + 16];

    v8f acc1[8];
#pragma unroll
    for (int t = 0; t < 8; ++t) {          // 8 independent WMMAs, no RAW stalls
      const int h0 = t * 16;
      V16 aw1;
      aw1.h[0] = *(const v8h*)&w1t[h0 + m][kb];
      aw1.h[1] = *(const v8h*)&w1t[h0 + m][kb + 16];
      v8f acc;
#pragma unroll
      for (int r = 0; r < 8; ++r) acc[r] = b1s[h0 + rbase + r];  // bias on M
      acc1[t] = __builtin_amdgcn_wmma_f32_16x16x32_f16(
          false, aw1.v, false, bedge.v, (short)0, acc, false, false);
    }
#pragma unroll
    for (int t = 0; t < 8; ++t) {          // relu + pack + single b128 store
      H8 hh;
#pragma unroll
      for (int r = 0; r < 8; ++r) hh.e[r] = (_Float16)fmaxf(acc1[t][r], 0.0f);
      *(v8h*)&h1s[w][m][t * 16 + rbase] = hh.v;
    }
    wave_lds_sync();

    // GEMM2: h2 = h1 @ W2 (M = neighbors); mean over M is cheap.
    // A-fragments (h1 K-chunks) are invariant across N-tiles: hoist them.
    V16 a2f[4];
#pragma unroll
    for (int c = 0; c < 4; ++c) {
      a2f[c].h[0] = *(const v8h*)&h1s[w][m][c * 32 + kb];
      a2f[c].h[1] = *(const v8h*)&h1s[w][m][c * 32 + kb + 16];
    }
#pragma unroll
    for (int tp = 0; tp < 8; ++tp) {       // two interleaved tiles per pass
      const int n0 = tp * 32;
      const float bva = b2s[n0 + m];
      const float bvb = b2s[n0 + 16 + m];
      v8f accA = {bva, bva, bva, bva, bva, bva, bva, bva};
      v8f accB = {bvb, bvb, bvb, bvb, bvb, bvb, bvb, bvb};
#pragma unroll
      for (int c = 0; c < 4; ++c) {
        V16 bfA, bfB;
        const _Float16* wpA = w2t + (size_t)(n0 + m) * H_ + c * 32 + kb;
        const _Float16* wpB = wpA + (size_t)16 * H_;
        bfA.h[0] = *(const v8h*)wpA; bfA.h[1] = *(const v8h*)(wpA + 16);
        bfB.h[0] = *(const v8h*)wpB; bfB.h[1] = *(const v8h*)(wpB + 16);
        accA = __builtin_amdgcn_wmma_f32_16x16x32_f16(
            false, a2f[c].v, false, bfA.v, (short)0, accA, false, false);
        accB = __builtin_amdgcn_wmma_f32_16x16x32_f16(
            false, a2f[c].v, false, bfB.v, (short)0, accB, false, false);
      }
      float sA = 0.0f, sB = 0.0f;
#pragma unroll
      for (int r = 0; r < 8; ++r) {
        sA += fmaxf(accA[r], 0.0f);
        sB += fmaxf(accB[r], 0.0f);
      }
      sA += __shfl_xor(sA, 16, 32);        // combine M=0..7 and M=8..15 halves
      sB += __shfl_xor(sB, 16, 32);
      if (lane < 16) {
        out[obase + n0 + lane]      = sA * (1.0f / 16.0f) * maskp;
        out[obase + n0 + 16 + lane] = sB * (1.0f / 16.0f) * maskp;
      }
    }
    if (lane == 0)
      out[obase + O_] = maskp;             // ones-channel * mask
  }
}

extern "C" void kernel_launch(void* const* d_in, const int* in_sizes, int n_in,
                              void* d_out, int out_size, void* d_ws, size_t ws_size,
                              hipStream_t stream) {
  (void)in_sizes; (void)n_in; (void)out_size; (void)ws_size;
  const float* x  = (const float*)d_in[0];
  const float* W1 = (const float*)d_in[1];
  const float* b1 = (const float*)d_in[2];
  const float* W2 = (const float*)d_in[3];
  const float* b2 = (const float*)d_in[4];
  _Float16* w2t = (_Float16*)d_ws;   // 64 KB: W2^T in f16

  prep_w2t_kernel<<<(H_ * O_ + 255) / 256, 256, 0, stream>>>(W2, w2t);
  edgeconv_kernel<<<B_ * 2, 256, 0, stream>>>(x, W1, b1, b2, w2t, (float*)d_out);
}
